// SubgraphGINEncoder_10634339025391
// MI455X (gfx1250) — matrix operand order repaired and checked by reading, not compiled, not run on hardware
//
#include <hip/hip_runtime.h>

typedef __attribute__((ext_vector_type(16))) __bf16 v16bf;
typedef __attribute__((ext_vector_type(8)))  __bf16 v8bf;
typedef __attribute__((ext_vector_type(8)))  float  v8f;

#define NF   524288   // 8192 * 64 flat nodes
#define HDIM 64
#define FIN  128
#define NB   8192
#define NK   64
#define NG   512

static __device__ __forceinline__ v16bf cat16(v8bf lo, v8bf hi) {
  v16bf r;
#pragma unroll
  for (int i = 0; i < 8; ++i) { r[i] = lo[i]; r[i + 8] = hi[i]; }
  return r;
}

// ---------------- Kernel 1: gather + input projection (128 -> 64) + relu ----
__global__ void __launch_bounds__(256)
k_gather_proj(const float* __restrict__ xg, const int* __restrict__ nodes,
              const float* __restrict__ Wp, const float* __restrict__ bp,
              float* __restrict__ h)
{
  __shared__ __align__(16) __bf16 wt[HDIM * FIN];   // transposed: wt[n*128 + k]
  __shared__ __align__(16) __bf16 xl[8][16 * FIN];  // per-wave 16x128 tile
  const int tid = threadIdx.x, wid = tid >> 5, lane = tid & 31;

  for (int i = tid; i < FIN * HDIM; i += 256) {     // Wp is (128,64): Wp[k][n]
    int kk = i >> 6, nn = i & 63;
    wt[nn * FIN + kk] = (__bf16)Wp[i];
  }
  const int row0 = (blockIdx.x * 8 + wid) * 16;
  {
    int r = lane >> 1, cb = (lane & 1) * 64;
    int fid = nodes[row0 + r];
    bool valid = fid >= 0;
    const float4* src = (const float4*)(xg + (size_t)(valid ? fid : 0) * FIN + cb);
#pragma unroll
    for (int j = 0; j < 16; ++j) {
      float4 v = valid ? src[j] : make_float4(0.f, 0.f, 0.f, 0.f);
      __bf16* d = &xl[wid][r * FIN + cb + j * 4];
      d[0] = (__bf16)v.x; d[1] = (__bf16)v.y; d[2] = (__bf16)v.z; d[3] = (__bf16)v.w;
    }
  }
  __syncthreads();

  const int g = lane >> 4, mn = lane & 15;
  v8f acc[4];
#pragma unroll
  for (int nt = 0; nt < 4; ++nt) {
    float bv = bp[nt * 16 + mn];
#pragma unroll
    for (int r = 0; r < 8; ++r) acc[nt][r] = bv;
  }
#pragma unroll
  for (int kh = 0; kh < 4; ++kh) {
    v16bf a = cat16(*(const v8bf*)&xl[wid][mn * FIN + kh * 32 + 8 * g],
                    *(const v8bf*)&xl[wid][mn * FIN + kh * 32 + 8 * g + 16]);
#pragma unroll
    for (int nt = 0; nt < 4; ++nt) {
      v16bf b = cat16(*(const v8bf*)&wt[(nt * 16 + mn) * FIN + kh * 32 + 8 * g],
                      *(const v8bf*)&wt[(nt * 16 + mn) * FIN + kh * 32 + 8 * g + 16]);
      acc[nt] = __builtin_amdgcn_wmma_f32_16x16x32_bf16(false, a, false, b,
                                                        (short)0, acc[nt], false, false);
    }
  }
#pragma unroll
  for (int nt = 0; nt < 4; ++nt)
#pragma unroll
    for (int r = 0; r < 8; ++r) {
      float v = acc[nt][r]; v = v > 0.f ? v : 0.f;
      h[(size_t)(row0 + r + 8 * g) * HDIM + nt * 16 + mn] = v;
    }
}

// ---------------- Kernel 2: edge scatter-add (segment_sum over edges) -------
__global__ void __launch_bounds__(256)
k_edge_agg(const float* __restrict__ h, const int* __restrict__ src,
           const int* __restrict__ dst, float* __restrict__ agg, int nE)
{
  long long t = (long long)blockIdx.x * 256 + threadIdx.x;
  int e = (int)(t >> 6);
  if (e >= nE) return;
  int c = (int)(t & 63);
  int s = src[e], d = dst[e];
  unsafeAtomicAdd(&agg[(size_t)d * HDIM + c], h[(size_t)s * HDIM + c]);
}

// ---------------- Kernel 3: fused two-GEMM MLP (GIN layer / final MLP) ------
// mode GIN:   X = xsrc + agg,          out = relu(relu(X@W1+b1)@W2+b2)
// mode FINAL: X = xsrc * 1/max(cnt,1), out =      relu(X@W1+b1)@W2+b2
__global__ void __launch_bounds__(256)
k_mlp2(const float* __restrict__ xsrc, const float* __restrict__ agg,
       const float* __restrict__ cnt,
       const float* __restrict__ W1, const float* __restrict__ b1,
       const float* __restrict__ W2, const float* __restrict__ b2,
       float* __restrict__ out, int outRelu)
{
  __shared__ __align__(16) __bf16 w1t[HDIM * HDIM];
  __shared__ __align__(16) __bf16 w2t[HDIM * HDIM];
  __shared__ __align__(16) __bf16 xl[8][16 * HDIM];
  __shared__ __align__(16) __bf16 zl[8][16 * HDIM];
  const int tid = threadIdx.x, wid = tid >> 5, lane = tid & 31;

  for (int i = tid; i < HDIM * HDIM; i += 256) {    // W is (64,64): W[k][n]
    int kk = i >> 6, nn = i & 63;
    w1t[nn * HDIM + kk] = (__bf16)W1[i];
    w2t[nn * HDIM + kk] = (__bf16)W2[i];
  }
  const int row0 = (blockIdx.x * 8 + wid) * 16;
  {
    int r = lane >> 1, cb = (lane & 1) * 32;
    size_t base = (size_t)(row0 + r) * HDIM + cb;
    float scale = 1.f;
    if (cnt) { float c = cnt[row0 + r]; scale = c > 0.f ? 1.f / c : 0.f; }
#pragma unroll
    for (int j = 0; j < 8; ++j) {
      float4 v = ((const float4*)(xsrc + base))[j];
      if (agg) {
        float4 a4 = ((const float4*)(agg + base))[j];
        v.x += a4.x; v.y += a4.y; v.z += a4.z; v.w += a4.w;
      }
      v.x *= scale; v.y *= scale; v.z *= scale; v.w *= scale;
      __bf16* d = &xl[wid][r * HDIM + cb + j * 4];
      d[0] = (__bf16)v.x; d[1] = (__bf16)v.y; d[2] = (__bf16)v.z; d[3] = (__bf16)v.w;
    }
  }
  __syncthreads();

  const int g = lane >> 4, mn = lane & 15;
  // ---- GEMM1: relu(X @ W1 + b1) ----
  v8f acc[4];
#pragma unroll
  for (int nt = 0; nt < 4; ++nt) {
    float bv = b1[nt * 16 + mn];
#pragma unroll
    for (int r = 0; r < 8; ++r) acc[nt][r] = bv;
  }
#pragma unroll
  for (int kh = 0; kh < 2; ++kh) {
    v16bf a = cat16(*(const v8bf*)&xl[wid][mn * HDIM + kh * 32 + 8 * g],
                    *(const v8bf*)&xl[wid][mn * HDIM + kh * 32 + 8 * g + 16]);
#pragma unroll
    for (int nt = 0; nt < 4; ++nt) {
      v16bf b = cat16(*(const v8bf*)&w1t[(nt * 16 + mn) * HDIM + kh * 32 + 8 * g],
                      *(const v8bf*)&w1t[(nt * 16 + mn) * HDIM + kh * 32 + 8 * g + 16]);
      acc[nt] = __builtin_amdgcn_wmma_f32_16x16x32_bf16(false, a, false, b,
                                                        (short)0, acc[nt], false, false);
    }
  }
#pragma unroll
  for (int nt = 0; nt < 4; ++nt)
#pragma unroll
    for (int r = 0; r < 8; ++r) {
      float v = acc[nt][r]; v = v > 0.f ? v : 0.f;
      zl[wid][(r + 8 * g) * HDIM + nt * 16 + mn] = (__bf16)v;  // wave-local, in-order DS
    }
  // ---- GEMM2: Z @ W2 + b2 (+ optional relu) ----
  v8f acc2[4];
#pragma unroll
  for (int nt = 0; nt < 4; ++nt) {
    float bv = b2[nt * 16 + mn];
#pragma unroll
    for (int r = 0; r < 8; ++r) acc2[nt][r] = bv;
  }
#pragma unroll
  for (int kh = 0; kh < 2; ++kh) {
    v16bf a = cat16(*(const v8bf*)&zl[wid][mn * HDIM + kh * 32 + 8 * g],
                    *(const v8bf*)&zl[wid][mn * HDIM + kh * 32 + 8 * g + 16]);
#pragma unroll
    for (int nt = 0; nt < 4; ++nt) {
      v16bf b = cat16(*(const v8bf*)&w2t[(nt * 16 + mn) * HDIM + kh * 32 + 8 * g],
                      *(const v8bf*)&w2t[(nt * 16 + mn) * HDIM + kh * 32 + 8 * g + 16]);
      acc2[nt] = __builtin_amdgcn_wmma_f32_16x16x32_bf16(false, a, false, b,
                                                         (short)0, acc2[nt], false, false);
    }
  }
#pragma unroll
  for (int nt = 0; nt < 4; ++nt)
#pragma unroll
    for (int r = 0; r < 8; ++r) {
      float v = acc2[nt][r];
      if (outRelu) v = v > 0.f ? v : 0.f;
      out[(size_t)(row0 + r + 8 * g) * HDIM + nt * 16 + mn] = v;
    }
}

// ---------------- Kernel 4: subgraph mean pooling ---------------------------
__global__ void __launch_bounds__(256)
k_pool(const float* __restrict__ h, const int* __restrict__ nodes,
       float* __restrict__ reps, float* __restrict__ cnts)
{
  __shared__ float scnt[4];
  const int tid = threadIdx.x;
  const int b = blockIdx.x * 4 + (tid >> 6);
  const int c = tid & 63;
  if (c == 0) {
    int cc = 0;
#pragma unroll 4
    for (int i = 0; i < NK; ++i) cc += (nodes[b * NK + i] >= 0) ? 1 : 0;
    scnt[tid >> 6] = (float)cc;
    cnts[b] = (float)cc;
  }
  __syncthreads();
  float cc = scnt[tid >> 6];
  float s = 0.f;
#pragma unroll 8
  for (int i = 0; i < NK; ++i) s += h[((size_t)b * NK + i) * HDIM + c];
  reps[(size_t)b * HDIM + c] = cc > 0.f ? s / cc : 0.f;
}

// ---------------- Kernel 5: per-graph accumulation --------------------------
__global__ void __launch_bounds__(256)
k_gaccum(const float* __restrict__ reps, const float* __restrict__ cnts,
         const int* __restrict__ gid, float* __restrict__ psum,
         float* __restrict__ pcnt)
{
  int t = blockIdx.x * 256 + threadIdx.x;   // t < NB*64
  int b = t >> 6, c = t & 63;
  int g = gid[b];
  unsafeAtomicAdd(&psum[(size_t)g * HDIM + c], reps[(size_t)b * HDIM + c]);
  if (c == 0) unsafeAtomicAdd(&pcnt[g], cnts[b] > 0.f ? 1.f : 0.f);
}

// ---------------------------------------------------------------------------
extern "C" void kernel_launch(void* const* d_in, const int* in_sizes, int n_in,
                              void* d_out, int out_size, void* d_ws, size_t ws_size,
                              hipStream_t stream)
{
  const float* xg   = (const float*)d_in[0];
  const int*   nodes= (const int*)d_in[1];
  const int*   eidx = (const int*)d_in[2];
  const int*   gid  = (const int*)d_in[4];   // d_in[3]=edge_ptr (unused), d_in[5]=k
  const float* Wp   = (const float*)d_in[6];
  const float* bp   = (const float*)d_in[7];
  const float* gW1  = (const float*)d_in[8];
  const float* gb1  = (const float*)d_in[9];
  const float* gW2  = (const float*)d_in[10];
  const float* gb2  = (const float*)d_in[11];
  const float* fw1  = (const float*)d_in[12];
  const float* fb1  = (const float*)d_in[13];
  const float* fw2  = (const float*)d_in[14];
  const float* fb2  = (const float*)d_in[15];

  const int nE = in_sizes[2] / 2;
  const int* esrc = eidx;
  const int* edst = eidx + nE;

  char*  ws   = (char*)d_ws;
  float* h    = (float*)ws;                               // NF*64 f32
  float* agg  = (float*)(ws + (size_t)NF * HDIM * 4);     // NF*64 f32
  float* cnts = (float*)(ws + (size_t)NF * HDIM * 8);     // NB
  float* psum = cnts + NB;                                // NG*64
  float* pcnt = psum + NG * HDIM;                         // NG

  float* out_pg  = (float*)d_out;        // per_graph (512,64)
  float* out_rep = out_pg + NG * HDIM;   // subgraph_reps (8192,64)

  // h0 = relu(gather(x) @ Wp + bp)
  k_gather_proj<<<NF / 128, 256, 0, stream>>>(xg, nodes, Wp, bp, h);

  // 3 GIN layers
  for (int l = 0; l < 3; ++l) {
    hipMemsetAsync(agg, 0, (size_t)NF * HDIM * 4, stream);
    long long nthr = (long long)nE * 64;
    k_edge_agg<<<(int)((nthr + 255) / 256), 256, 0, stream>>>(h, esrc, edst, agg, nE);
    k_mlp2<<<NF / 128, 256, 0, stream>>>(h, agg, nullptr,
        gW1 + (size_t)l * HDIM * HDIM, gb1 + l * HDIM,
        gW2 + (size_t)l * HDIM * HDIM, gb2 + l * HDIM,
        h, /*outRelu=*/1);  // in place: row-disjoint per block
  }

  // subgraph mean pooling -> d_out reps + counts
  k_pool<<<NB / 4, 256, 0, stream>>>(h, nodes, out_rep, cnts);

  // per-graph mean
  hipMemsetAsync(psum, 0, (size_t)(NG * HDIM + NG) * 4, stream);
  k_gaccum<<<NB * HDIM / 256, 256, 0, stream>>>(out_rep, cnts, gid, psum, pcnt);

  // final MLP: relu(pg @ fw1 + fb1) @ fw2 + fb2  (with pg = psum/cnt)
  k_mlp2<<<NG / 128, 256, 0, stream>>>(psum, nullptr, pcnt,
      fw1, fb1, fw2, fb2, out_pg, /*outRelu=*/0);
}
